// Model_39676907885326
// MI455X (gfx1250) — matrix operand order repaired and checked by reading, not compiled, not run on hardware
//
#include <hip/hip_runtime.h>
#include <hip/hip_bf16.h>
#include <math.h>

typedef __bf16 bf16;
typedef __attribute__((ext_vector_type(16))) __bf16 v16bf;
typedef __attribute__((ext_vector_type(8)))  __bf16 bf16x8;
typedef __attribute__((ext_vector_type(4)))  __bf16 bf16x4;
typedef __attribute__((ext_vector_type(8)))  float  v8f;

constexpr int   kN    = 384;
constexpr int   kNN   = kN * kN;
constexpr float kDrop = 0.5f;
#define LDT 40   // padded LDS row stride (bf16 elements)

// ---------------------------------------------------------------------------
// Fragment loaders matching CDNA5 WMMA 16x16x32 bf16 VGPR layouts (wave32).
// A tile in LDS: row-major [row][LDT], K-chunk of 32 at col 0.
// B tile in LDS: row-major [n][LDT] over K, so per-lane reads are contiguous.
// ---------------------------------------------------------------------------
__device__ __forceinline__ v16bf frag_a(const bf16* As, int row0) {
    const int lane = threadIdx.x & 31;
    const bf16* p = As + (row0 + (lane & 15)) * LDT + ((lane >> 4) << 3);
    bf16x8 lo = *(const bf16x8*)p;          // K = kb+0..7
    bf16x8 hi = *(const bf16x8*)(p + 16);   // K = kb+16..23
    return __builtin_shufflevector(lo, hi, 0,1,2,3,4,5,6,7,8,9,10,11,12,13,14,15);
}

__device__ __forceinline__ v16bf frag_b(const bf16* Bs, int col0) {
    const int lane = threadIdx.x & 31;
    const bf16* p = Bs + (col0 + (lane & 15)) * LDT + ((lane >> 4) << 3);
    bf16x8 lo = *(const bf16x8*)p;          // K = kb+0..7
    bf16x8 hi = *(const bf16x8*)(p + 16);   // K = kb+16..23
    return __builtin_shufflevector(lo, hi, 0,1,2,3,4,5,6,7,8,9,10,11,12,13,14,15);
}

// ---------------------------------------------------------------------------
// Pass 0: wt[j,l,k] = bf16(w[l,k,j])  -- (j,k) 32x32 tiles via LDS, per l.
// ---------------------------------------------------------------------------
__global__ __launch_bounds__(256) void wt_transpose_kernel(const float* __restrict__ w,
                                                           bf16* __restrict__ wt) {
    __shared__ float tile[32][33];
    const int j0 = blockIdx.x * 32;
    const int k0 = blockIdx.y * 32;
    const int lb = blockIdx.z * 16;
    const int tx = threadIdx.x;   // 0..31
    const int ty = threadIdx.y;   // 0..7
    for (int ll = 0; ll < 16; ++ll) {
        const int l = lb + ll;
#pragma unroll
        for (int s = 0; s < 4; ++s) {
            const int kr = ty + 8 * s;
            tile[kr][tx] = w[(size_t)l * kNN + (size_t)(k0 + kr) * kN + (j0 + tx)];
        }
        __syncthreads();
#pragma unroll
        for (int s = 0; s < 4; ++s) {
            const int jr = ty + 8 * s;
            wt[(size_t)(j0 + jr) * kNN + (size_t)l * kN + (k0 + tx)] = (bf16)tile[tx][jr];
        }
        __syncthreads();
    }
}

// ---------------------------------------------------------------------------
// Pass 1: per-row softmax stats.  row = i*N + j.
// ---------------------------------------------------------------------------
__global__ __launch_bounds__(256) void stats_kernel(const float* __restrict__ inp,
                                                    const float* __restrict__ w,
                                                    float* __restrict__ rowC,
                                                    float* __restrict__ rowM,
                                                    float* __restrict__ rowZ) {
    const int row  = blockIdx.x * 8 + (threadIdx.x >> 5);
    const int lane = threadIdx.x & 31;
    const int j    = row % kN;
    const float* ip = inp + (size_t)row * kN;
    const float* wp = w   + (size_t)row * kN;

    float s = 0.f, sw = 0.f;
    for (int l = lane; l < kN; l += 32) { s += ip[l]; sw += wp[l]; }
#pragma unroll
    for (int off = 16; off > 0; off >>= 1) {
        s  += __shfl_xor(s,  off);
        sw += __shfl_xor(sw, off);
    }
    const float c1 = s / (kDrop + sw);

    float m = -INFINITY;
    for (int l = lane; l <= j; l += 32) m = fmaxf(m, c1 * wp[l]);
#pragma unroll
    for (int off = 16; off > 0; off >>= 1) m = fmaxf(m, __shfl_xor(m, off));

    float z = 0.f;
    for (int l = lane; l <= j; l += 32) z += __expf(c1 * wp[l] - m);
#pragma unroll
    for (int off = 16; off > 0; off >>= 1) z += __shfl_xor(z, off);

    if (lane == 0) { rowC[row] = c1; rowM[row] = m; rowZ[row] = 1.0f / z; }
}

// ---------------------------------------------------------------------------
// Pass 2: E[:,j,:] = input[:,j,:] x wt[j]  (one 384^3 GEMM per j)
// 128x128 block tile, 8 waves of 32x64, K=32 steps, bf16 WMMA.
// Register-staged double-buffered LDS: one barrier per K-step.
// ---------------------------------------------------------------------------
__global__ __launch_bounds__(256) void e_gemm_kernel(const float* __restrict__ inp,
                                                     const bf16* __restrict__ wt,
                                                     bf16* __restrict__ Ebf) {
    __shared__ bf16 As[2][128 * LDT];   // [i_local][k]
    __shared__ bf16 Bs[2][128 * LDT];   // [l_local][k]
    const int jj = blockIdx.y;
    const int i0 = (blockIdx.x / 3) * 128;
    const int l0 = (blockIdx.x % 3) * 128;
    const int tid  = threadIdx.x;
    const int wave = tid >> 5;
    const int wrow = (wave & 3) * 32;
    const int wcol = (wave >> 2) * 64;

    const float* abase = inp + (size_t)jj * kN;   // + i*kNN + k
    const bf16*  bbase = wt + (size_t)jj * kNN;   // + l*kN + k

    v8f c[2][4] = {};
    float4 ar[4];
    bf16x8 br[2];

    // ---- prologue: fill buffer 0 with k0 = 0 ----
#pragma unroll
    for (int s = 0; s < 4; ++s) {
        const int chunk = tid + 256 * s;
        const int r  = chunk >> 3;
        const int c4 = (chunk & 7) << 2;
        ar[s] = *(const float4*)(abase + (size_t)(i0 + r) * kNN + c4);
    }
#pragma unroll
    for (int s = 0; s < 2; ++s) {
        const int chunk = tid + 256 * s;
        const int r  = chunk >> 2;
        const int k8 = (chunk & 3) << 3;
        br[s] = *(const bf16x8*)(bbase + (size_t)(l0 + r) * kN + k8);
    }
#pragma unroll
    for (int s = 0; s < 4; ++s) {
        const int chunk = tid + 256 * s;
        const int r  = chunk >> 3;
        const int c4 = (chunk & 7) << 2;
        bf16x4 o; o[0] = (bf16)ar[s].x; o[1] = (bf16)ar[s].y; o[2] = (bf16)ar[s].z; o[3] = (bf16)ar[s].w;
        *(bf16x4*)(&As[0][r * LDT + c4]) = o;
    }
#pragma unroll
    for (int s = 0; s < 2; ++s) {
        const int chunk = tid + 256 * s;
        const int r  = chunk >> 2;
        const int k8 = (chunk & 3) << 3;
        *(bf16x8*)(&Bs[0][r * LDT + k8]) = br[s];
    }
    __syncthreads();

    int p = 0;
    for (int k0 = 0; k0 < kN; k0 += 32) {
        const int kn = k0 + 32;
        const bool more = (kn < kN);
        // ---- stage next K-step's tiles into registers ----
        if (more) {
#pragma unroll
            for (int s = 0; s < 4; ++s) {
                const int chunk = tid + 256 * s;
                const int r  = chunk >> 3;
                const int c4 = (chunk & 7) << 2;
                ar[s] = *(const float4*)(abase + (size_t)(i0 + r) * kNN + kn + c4);
            }
#pragma unroll
            for (int s = 0; s < 2; ++s) {
                const int chunk = tid + 256 * s;
                const int r  = chunk >> 2;
                const int k8 = (chunk & 3) << 3;
                br[s] = *(const bf16x8*)(bbase + (size_t)(l0 + r) * kN + kn + k8);
            }
        }
        // ---- compute from current buffer ----
        const bf16* Ap = As[p];
        const bf16* Bp = Bs[p];
        const v16bf a0 = frag_a(Ap, wrow);
        const v16bf a1 = frag_a(Ap, wrow + 16);
#pragma unroll
        for (int fc = 0; fc < 4; ++fc) {
            const v16bf b = frag_b(Bp, wcol + fc * 16);
            c[0][fc] = __builtin_amdgcn_wmma_f32_16x16x32_bf16(false, a0, false, b, (short)0, c[0][fc], false, false);
            c[1][fc] = __builtin_amdgcn_wmma_f32_16x16x32_bf16(false, a1, false, b, (short)0, c[1][fc], false, false);
        }
        // ---- store staged registers into the other buffer ----
        if (more) {
            bf16* An = As[p ^ 1];
            bf16* Bn = Bs[p ^ 1];
#pragma unroll
            for (int s = 0; s < 4; ++s) {
                const int chunk = tid + 256 * s;
                const int r  = chunk >> 3;
                const int c4 = (chunk & 7) << 2;
                bf16x4 o; o[0] = (bf16)ar[s].x; o[1] = (bf16)ar[s].y; o[2] = (bf16)ar[s].z; o[3] = (bf16)ar[s].w;
                *(bf16x4*)(&An[r * LDT + c4]) = o;
            }
#pragma unroll
            for (int s = 0; s < 2; ++s) {
                const int chunk = tid + 256 * s;
                const int r  = chunk >> 2;
                const int k8 = (chunk & 3) << 3;
                *(bf16x8*)(&Bn[r * LDT + k8]) = br[s];
            }
            __syncthreads();
            p ^= 1;
        }
    }

    // Store D -> Ebf[i, jj, l] (bf16)
    const int lane = tid & 31;
    const int nloc = lane & 15;
    const int roff = (lane >> 4) * 8;
#pragma unroll
    for (int fr = 0; fr < 2; ++fr)
#pragma unroll
        for (int fc = 0; fc < 4; ++fc)
#pragma unroll
            for (int v = 0; v < 8; ++v) {
                const int grow = i0 + wrow + fr * 16 + roff + v;   // i
                const int gcol = l0 + wcol + fc * 16 + nloc;       // l
                Ebf[(size_t)grow * kNN + (size_t)jj * kN + gcol] = (bf16)c[fr][fc][v];
            }
}

// ---------------------------------------------------------------------------
// Pass 2.5: Et[i,m,l] = Ebf[i,l,m]  (transpose last two dims, 32x32 LDS tiles)
// ---------------------------------------------------------------------------
__global__ __launch_bounds__(256) void et_transpose_kernel(const bf16* __restrict__ Ebf,
                                                           bf16* __restrict__ Et) {
    __shared__ bf16 tile[32 * 36];
    const int l0 = blockIdx.x * 32;
    const int m0 = blockIdx.y * 32;
    const int i  = blockIdx.z;
    const int tid = threadIdx.x;
    const int r  = tid >> 3;          // 0..31
    const int c4 = (tid & 7) << 2;    // 0..28

    const bf16x4 v = *(const bf16x4*)(Ebf + (size_t)i * kNN + (size_t)(l0 + r) * kN + m0 + c4);
#pragma unroll
    for (int e = 0; e < 4; ++e) tile[r * 36 + c4 + e] = v[e];
    __syncthreads();

    bf16x4 o;
#pragma unroll
    for (int e = 0; e < 4; ++e) o[e] = tile[(c4 + e) * 36 + r];
    *(bf16x4*)(Et + (size_t)i * kNN + (size_t)(m0 + r) * kN + l0 + c4) = o;
}

// ---------------------------------------------------------------------------
// Pass 3: out[i] = attn[i] @ E[i], attn built on the fly from w + row stats.
// Same register-staged double-buffered structure as pass 2.
// ---------------------------------------------------------------------------
__global__ __launch_bounds__(256) void attn_out_kernel(const float* __restrict__ w,
                                                       const bf16* __restrict__ Et,
                                                       const float* __restrict__ rowC,
                                                       const float* __restrict__ rowM,
                                                       const float* __restrict__ rowZ,
                                                       float* __restrict__ out) {
    __shared__ bf16  Ps[2][128 * LDT];   // attn tile [j_local][l]
    __shared__ bf16  Es[2][128 * LDT];   // E tile    [m_local][l]
    __shared__ float sC[128], sM[128], sZ[128];
    const int i  = blockIdx.y;
    const int j0 = (blockIdx.x / 3) * 128;
    const int m0 = (blockIdx.x % 3) * 128;
    const int tid  = threadIdx.x;
    const int wave = tid >> 5;
    const int wrow = (wave & 3) * 32;
    const int wcol = (wave >> 2) * 64;

    if (tid < 128) {
        const int rr = i * kN + j0 + tid;
        sC[tid] = rowC[rr]; sM[tid] = rowM[rr]; sZ[tid] = rowZ[rr];
    }
    __syncthreads();

    const int   pr = tid >> 1;          // 0..127: attn row handled by this thread
    const int   ph = (tid & 1) * 16;    // half of the 32-wide l chunk
    const float c1 = sC[pr];
    const float mr = sM[pr];
    const float iz = sZ[pr];
    const int   jg = j0 + pr;

    const float* wrowp = w + (size_t)i * kNN + (size_t)jg * kN;   // + l
    const bf16*  ebase = Et + (size_t)i * kNN;                    // + m*kN + l

    v8f c[2][4] = {};
    float4 wr[4];
    bf16x8 er[2];

    // ---- helper expansion: fill buffer `buf` from staged regs for chunk base lb ----
    // (written inline below; prologue fills buffer 0 with l-chunk 0)
#pragma unroll
    for (int s = 0; s < 4; ++s)
        wr[s] = *(const float4*)(wrowp + ph + s * 4);
#pragma unroll
    for (int s = 0; s < 2; ++s) {
        const int chunk = tid + 256 * s;
        const int r  = chunk >> 2;
        const int k8 = (chunk & 3) << 3;
        er[s] = *(const bf16x8*)(ebase + (size_t)(m0 + r) * kN + k8);
    }
#pragma unroll
    for (int s = 0; s < 4; ++s) {
        const int cc = ph + s * 4;
        float vv[4] = { wr[s].x, wr[s].y, wr[s].z, wr[s].w };
        bf16x4 o;
#pragma unroll
        for (int e = 0; e < 4; ++e) {
            const int lg = cc + e;
            const float pv = (lg <= jg) ? __expf(c1 * vv[e] - mr) * iz : 0.0f;
            o[e] = (bf16)pv;
        }
        *(bf16x4*)(&Ps[0][pr * LDT + cc]) = o;
    }
#pragma unroll
    for (int s = 0; s < 2; ++s) {
        const int chunk = tid + 256 * s;
        const int r  = chunk >> 2;
        const int k8 = (chunk & 3) << 3;
        *(bf16x8*)(&Es[0][r * LDT + k8]) = er[s];
    }
    __syncthreads();

    int p = 0;
    for (int l0 = 0; l0 < kN; l0 += 32) {
        const int ln = l0 + 32;
        const bool more = (ln < kN);
        // ---- stage next chunk ----
        if (more) {
#pragma unroll
            for (int s = 0; s < 4; ++s)
                wr[s] = *(const float4*)(wrowp + ln + ph + s * 4);
#pragma unroll
            for (int s = 0; s < 2; ++s) {
                const int chunk = tid + 256 * s;
                const int r  = chunk >> 2;
                const int k8 = (chunk & 3) << 3;
                er[s] = *(const bf16x8*)(ebase + (size_t)(m0 + r) * kN + ln + k8);
            }
        }
        // ---- compute from current buffer ----
        const bf16* Pp = Ps[p];
        const bf16* Ep = Es[p];
        const v16bf a0 = frag_a(Pp, wrow);
        const v16bf a1 = frag_a(Pp, wrow + 16);
#pragma unroll
        for (int fc = 0; fc < 4; ++fc) {
            const v16bf b = frag_b(Ep, wcol + fc * 16);
            c[0][fc] = __builtin_amdgcn_wmma_f32_16x16x32_bf16(false, a0, false, b, (short)0, c[0][fc], false, false);
            c[1][fc] = __builtin_amdgcn_wmma_f32_16x16x32_bf16(false, a1, false, b, (short)0, c[1][fc], false, false);
        }
        // ---- convert + store staged regs into the other buffer ----
        if (more) {
            bf16* Pn = Ps[p ^ 1];
            bf16* En = Es[p ^ 1];
#pragma unroll
            for (int s = 0; s < 4; ++s) {
                const int cc = ph + s * 4;
                float vv[4] = { wr[s].x, wr[s].y, wr[s].z, wr[s].w };
                bf16x4 o;
#pragma unroll
                for (int e = 0; e < 4; ++e) {
                    const int lg = ln + cc + e;
                    const float pv = (lg <= jg) ? __expf(c1 * vv[e] - mr) * iz : 0.0f;
                    o[e] = (bf16)pv;
                }
                *(bf16x4*)(&Pn[pr * LDT + cc]) = o;
            }
#pragma unroll
            for (int s = 0; s < 2; ++s) {
                const int chunk = tid + 256 * s;
                const int r  = chunk >> 2;
                const int k8 = (chunk & 3) << 3;
                *(bf16x8*)(&En[r * LDT + k8]) = er[s];
            }
            __syncthreads();
            p ^= 1;
        }
    }

    // Store D -> out[i, j, m] (f32, 64B contiguous per instruction)
    const int lane = tid & 31;
    const int nloc = lane & 15;
    const int roff = (lane >> 4) * 8;
#pragma unroll
    for (int fr = 0; fr < 2; ++fr)
#pragma unroll
        for (int fc = 0; fc < 4; ++fc)
#pragma unroll
            for (int v = 0; v < 8; ++v) {
                const int grow = j0 + wrow + fr * 16 + roff + v;   // j
                const int gcol = m0 + wcol + fc * 16 + nloc;       // m
                out[(size_t)i * kNN + (size_t)grow * kN + gcol] = c[fr][fc][v];
            }
}

// ---------------------------------------------------------------------------
extern "C" void kernel_launch(void* const* d_in, const int* in_sizes, int n_in,
                              void* d_out, int out_size, void* d_ws, size_t ws_size,
                              hipStream_t stream) {
    (void)in_sizes; (void)n_in; (void)out_size; (void)ws_size;
    const float* inp = (const float*)d_in[0];
    const float* w   = (const float*)d_in[1];
    float*       out = (float*)d_out;

    const size_t NNN = (size_t)kN * kNN;
    bf16*  wt    = (bf16*)d_ws;          // 113 MB; reused as Et after pass 2
    bf16*  Ebf   = wt + NNN;             // 113 MB
    bf16*  Et    = wt;                   // alias: wt is dead after e_gemm
    float* stats = (float*)(Ebf + NNN);  // 3 * N^2 f32
    float* rowC  = stats;
    float* rowM  = stats + (size_t)kNN;
    float* rowZ  = stats + 2 * (size_t)kNN;

    wt_transpose_kernel<<<dim3(12, 12, 24), dim3(32, 8), 0, stream>>>(w, wt);
    stats_kernel<<<dim3(kNN / 8), dim3(256), 0, stream>>>(inp, w, rowC, rowM, rowZ);
    e_gemm_kernel<<<dim3(9, kN), dim3(256), 0, stream>>>(inp, wt, Ebf);
    et_transpose_kernel<<<dim3(12, 12, kN), dim3(256), 0, stream>>>(Ebf, Et);
    attn_out_kernel<<<dim3(9, kN), dim3(256), 0, stream>>>(w, Et, rowC, rowM, rowZ, out);
}
